// CategoryLevelContrastive_69561290326312
// MI455X (gfx1250) — matrix operand order repaired and checked by reading, not compiled, not run on hardware
//
#include <hip/hip_runtime.h>
#include <hip/hip_bf16.h>

// ---------------------------------------------------------------------------
// CategoryLevelContrastive fused pipeline for gfx1250 (MI455X, wave32, WMMA)
//   K0: zero class accumulators in workspace
//   K1: fused  proj(GEMM1+ReLU+GEMM2, bf16 WMMA, b128 LDS frags, prefetch,
//       sched_group_barrier-pipelined DS->WMMA) -> L2norm -> pdf(exp kernel)
//       -> softmax -> per-class LDS accumulation -> global atomic flush
//   K2: finalize  dA/dB/M -> KL matrix J(32x32) -> contrastive loss scalar
// ---------------------------------------------------------------------------

typedef unsigned int   u32;
typedef unsigned short u16;
typedef __bf16 v16bf __attribute__((ext_vector_type(16)));
typedef float  v8f   __attribute__((ext_vector_type(8)));

union Frag { uint4 q[2]; v16bf v; };   // 32 bytes = 8 VGPRs

static __device__ __forceinline__ u16 f2bf(float f) {
    union { float f; u32 u; } x; x.f = f;
    u32 r = x.u + 0x7FFFu + ((x.u >> 16) & 1u);   // round-to-nearest-even
    return (u16)(r >> 16);
}
static __device__ __forceinline__ u32 f2bf2(float a, float b) {
    return (u32)f2bf(a) | ((u32)f2bf(b) << 16);
}

// DS-read / WMMA software pipeline for one kk-step: 2 A-loads + 2 B0-loads
// first, then (2 B-loads, 1 WMMA) x7, final WMMA. Masks: 0x100=DS-read,
// 0x008=MFMA/WMMA.
static __device__ __forceinline__ void sched_pipeline_kk() {
    __builtin_amdgcn_sched_group_barrier(0x100, 4, 0);
#pragma unroll
    for (int t = 0; t < 7; ++t) {
        __builtin_amdgcn_sched_group_barrier(0x100, 2, 0);
        __builtin_amdgcn_sched_group_barrier(0x008, 1, 0);
    }
    __builtin_amdgcn_sched_group_barrier(0x008, 1, 0);
}

#define NROWS 128      // rows per block
#define INF_  512      // IN_FEATURES
#define OUTF  128      // OUT_FEATURES
#define NBINS 128
#define NCLS  16
#define KC    64       // K-chunk for GEMM1
#define SW1   72       // w1t stride (ushorts), 144 B: 16B-aligned rows
#define SW2   136      // w2t stride (ushorts), 272 B: 16B-aligned rows

// ---- dynamic LDS layout (bytes) -------------------------------------------
// zbuf  f32 [128][132]          :      0 .. 67584
// pdf   f32 [128][129]          :  67584 .. 133632
// bins  f32 [128]               : 133632 .. 134144
// accL  f32 [16][128]           : 134144 .. 142336
// cntL  f32 [16]                : 142336 .. 142400
// aliased into zbuf region (phase-disjoint, barrier-separated):
//   xt  u16 [128][64]   @ 0       (GEMM1 A)
//   w1t u16 [128][72]   @ 16384   (GEMM1 B, stored [n][k])   -> 34816
//   ht  u16 [128][128]  @ 0       (GEMM2 A)                  -> 32768
//   w2t u16 [128][136]  @ 32768   (GEMM2 B, stored [n][k])   -> 67584
#define ZB_OFF   0
#define PDF_OFF  67584
#define BIN_OFF  133632
#define ACC_OFF  134144
#define CNT_OFF  142336
#define DYN_LDS  142400
#define XT_OFF   0
#define W1T_OFF  16384
#define HT_OFF   0
#define W2T_OFF  32768

__global__ void clc_fused_kernel(const float* __restrict__ x1,
                                 const int*   __restrict__ y1,
                                 const float* __restrict__ x2,
                                 const int*   __restrict__ y2,
                                 const float* __restrict__ W1,
                                 const float* __restrict__ b1,
                                 const float* __restrict__ W2,
                                 const float* __restrict__ b2,
                                 const float* __restrict__ bins,
                                 float* __restrict__ gsums,   // [2][16][128]
                                 float* __restrict__ gcnt)    // [2][16]
{
    extern __shared__ char smem[];
    float* zb   = (float*)(smem + ZB_OFF);    // stride 132
    float* pdf  = (float*)(smem + PDF_OFF);   // stride 129
    float* binsS= (float*)(smem + BIN_OFF);
    float* accL = (float*)(smem + ACC_OFF);
    float* cntL = (float*)(smem + CNT_OFF);
    u16*   xt   = (u16*)(smem + XT_OFF);      // stride 64
    u16*   w1t  = (u16*)(smem + W1T_OFF);     // stride SW1
    u16*   ht   = (u16*)(smem + HT_OFF);      // stride 128
    u16*   w2t  = (u16*)(smem + W2T_OFF);     // stride SW2

    const int tid  = threadIdx.x;
    const int lane = tid & 31;
    const int wv   = tid >> 5;        // 0..7 (wave32)
    const int r16  = lane & 15;
    const int hf   = lane >> 4;       // lane-half per ISA fragment layouts

    const int input = blockIdx.x >> 5;     // 0 or 1
    const int rb    = blockIdx.x & 31;     // row block
    const int row0  = rb * NROWS;

    const float* X = (input == 0) ? x1 : x2;
    const int*   Y = (input == 0) ? y1 : y2;

    // ---- init: class accumulators + bins ----------------------------------
    for (int i = tid; i < NCLS * NBINS; i += 256) accL[i] = 0.f;
    if (tid < NCLS)  cntL[tid] = 0.f;
    if (tid < NBINS) binsS[tid] = bins[tid];

    // ---- GEMM1: h = relu(x @ W1 + b1), K=512 in chunks of 64 --------------
    v8f acc[8];
#pragma unroll
    for (int t = 0; t < 8; ++t) acc[t] = (v8f){0,0,0,0,0,0,0,0};

    for (int ck = 0; ck < INF_ / KC; ++ck) {
        const int k0 = ck * KC;
        __syncthreads();
        // x tile: float4 global loads -> packed bf16 ds_store_b64
        for (int idx = tid; idx < (NROWS * KC) / 4; idx += 256) {
            int r = idx >> 4, k4 = (idx & 15) * 4;
            float4 v = *(const float4*)&X[(row0 + r) * INF_ + k0 + k4];
            uint2 p; p.x = f2bf2(v.x, v.y); p.y = f2bf2(v.z, v.w);
            *(uint2*)&xt[r * 64 + k4] = p;
        }
        // W1 tile transposed: float4 over n, scalar bf16 LDS stores
        for (int idx = tid; idx < (KC * OUTF) / 4; idx += 256) {
            int k = idx >> 5, n4 = (idx & 31) * 4;
            float4 v = *(const float4*)&W1[(k0 + k) * OUTF + n4];
            w1t[(n4 + 0) * SW1 + k] = f2bf(v.x);
            w1t[(n4 + 1) * SW1 + k] = f2bf(v.y);
            w1t[(n4 + 2) * SW1 + k] = f2bf(v.z);
            w1t[(n4 + 3) * SW1 + k] = f2bf(v.w);
        }
        __syncthreads();
        // prefetch next chunk while the matrix cores chew on this one
        if (ck + 1 < INF_ / KC) {
            __builtin_prefetch(&X[(row0 + (tid >> 1)) * INF_ + k0 + KC + (tid & 1) * 32], 0, 0);
            __builtin_prefetch(&W1[(k0 + KC + (tid >> 2)) * OUTF + (tid & 3) * 32], 0, 0);
        }
#pragma unroll
        for (int kk = 0; kk < KC; kk += 32) {
            Frag A;
            const int ra = (wv * 16 + r16) * 64 + kk + hf * 8;
            A.q[0] = *(const uint4*)&xt[ra];
            A.q[1] = *(const uint4*)&xt[ra + 16];
#pragma unroll
            for (int t = 0; t < 8; ++t) {
                Frag B;
                const int rbn = (t * 16 + r16) * SW1 + kk + hf * 8;
                B.q[0] = *(const uint4*)&w1t[rbn];
                B.q[1] = *(const uint4*)&w1t[rbn + 16];
                acc[t] = __builtin_amdgcn_wmma_f32_16x16x32_bf16(
                    false, A.v, false, B.v, (short)0, acc[t], false, false);
            }
            sched_pipeline_kk();
        }
    }
    __syncthreads();

    // bias + relu -> bf16 h tile in LDS; stage W2 transposed
#pragma unroll
    for (int t = 0; t < 8; ++t) {
        const int n = t * 16 + r16;
        const float bb = b1[n];
#pragma unroll
        for (int j = 0; j < 8; ++j) {
            int row = wv * 16 + hf * 8 + j;
            float v = acc[t][j] + bb;
            ht[row * 128 + n] = f2bf(fmaxf(v, 0.f));
        }
    }
    for (int idx = tid; idx < (OUTF * OUTF) / 4; idx += 256) {
        int k = idx >> 5, n4 = (idx & 31) * 4;
        float4 v = *(const float4*)&W2[k * OUTF + n4];
        w2t[(n4 + 0) * SW2 + k] = f2bf(v.x);
        w2t[(n4 + 1) * SW2 + k] = f2bf(v.y);
        w2t[(n4 + 2) * SW2 + k] = f2bf(v.z);
        w2t[(n4 + 3) * SW2 + k] = f2bf(v.w);
    }
    __syncthreads();

    // ---- GEMM2: z = h @ W2 + b2, K=128 ------------------------------------
#pragma unroll
    for (int t = 0; t < 8; ++t) acc[t] = (v8f){0,0,0,0,0,0,0,0};
#pragma unroll
    for (int kk = 0; kk < OUTF; kk += 32) {
        Frag A;
        const int ra = (wv * 16 + r16) * 128 + kk + hf * 8;
        A.q[0] = *(const uint4*)&ht[ra];
        A.q[1] = *(const uint4*)&ht[ra + 16];
#pragma unroll
        for (int t = 0; t < 8; ++t) {
            Frag B;
            const int rbn = (t * 16 + r16) * SW2 + kk + hf * 8;
            B.q[0] = *(const uint4*)&w2t[rbn];
            B.q[1] = *(const uint4*)&w2t[rbn + 16];
            acc[t] = __builtin_amdgcn_wmma_f32_16x16x32_bf16(
                false, A.v, false, B.v, (short)0, acc[t], false, false);
        }
        sched_pipeline_kk();
    }
    __syncthreads();  // all h reads done before z overwrites the alias region

#pragma unroll
    for (int t = 0; t < 8; ++t) {
        const int n = t * 16 + r16;
        const float bb = b2[n];
#pragma unroll
        for (int j = 0; j < 8; ++j) {
            int row = wv * 16 + hf * 8 + j;
            zb[row * 132 + n] = acc[t][j] + bb;
        }
    }
    __syncthreads();

    // ---- row L2 normalize (stride 132 words) ---------
    if (tid < NROWS) {
        float s = 0.f;
        for (int d = 0; d < OUTF; ++d) { float v = zb[tid * 132 + d]; s += v * v; }
        float inv = 1.0f / sqrtf(s);
        for (int d = 0; d < OUTF; ++d) zb[tid * 132 + d] *= inv;
    }
    __syncthreads();

    // ---- marginal pdf: fused, no (b,D,bins) intermediate ever hits HBM ----
    // 16384 (row,bin) tasks; each sums 128 v_exp_f32. Lanes in a wave share
    // the row (broadcast LDS reads of x_norm).
    for (int q = 0; q < 64; ++q) {
        int task = tid + q * 256;
        int r = task >> 7, k = task & 127;
        float bk = binsS[k];
        float s = 0.f;
        for (int d = 0; d < OUTF; ++d) {
            float u = (zb[r * 132 + d] - bk) * 5.0f;   // 1/SIGMA = 5
            s += __expf(-0.5f * u * u);
        }
        pdf[r * 129 + k] = s * (1.0f / 128.0f);        // mean over D
    }
    __syncthreads();

    // ---- per-row normalize + softmax + per-class LDS accumulation ---------
    if (tid < NROWS) {
        const int base = tid * 129;
        float s1 = 0.f;
        for (int k = 0; k < NBINS; ++k) s1 += pdf[base + k];
        float inv1 = 1.0f / (s1 + 1e-8f);
        float mx = -1e30f;
        for (int k = 0; k < NBINS; ++k) mx = fmaxf(mx, pdf[base + k] * inv1);
        float s2 = 0.f;
        for (int k = 0; k < NBINS; ++k) s2 += __expf(pdf[base + k] * inv1 - mx);
        float invs2 = 1.0f / s2;
        int c = Y[row0 + tid]; if ((unsigned)c >= (unsigned)NCLS) c = 0;
        atomicAdd(&cntL[c], 1.0f);
        for (int k = 0; k < NBINS; ++k)
            atomicAdd(&accL[c * 128 + k], __expf(pdf[base + k] * inv1 - mx) * invs2);
    }
    __syncthreads();

    // ---- flush block-local class sums to global ---------------------------
    for (int idx = tid; idx < NCLS * NBINS; idx += 256)
        atomicAdd(&gsums[input * NCLS * NBINS + idx], accL[idx]);
    if (tid < NCLS)
        atomicAdd(&gcnt[input * NCLS + tid], cntL[tid]);
}

// ---------------------------------------------------------------------------
__global__ void clc_finalize_kernel(const float* __restrict__ gsums,
                                    const float* __restrict__ gcnt,
                                    float* __restrict__ out)
{
    __shared__ float D[48 * 129];   // rows: 0-15 dA, 16-31 dB, 32-47 M
    __shared__ float LG[48 * 129];
    __shared__ float Jm[32 * 33];
    __shared__ float partial[32];
    const int tid = threadIdx.x;

    for (int idx = tid; idx < NCLS * NBINS; idx += 256) {
        int c = idx >> 7, k = idx & 127;
        float a = gsums[idx] / gcnt[c];
        float b = gsums[NCLS * NBINS + idx] / gcnt[NCLS + c];
        D[c * 129 + k]        = a;
        D[(16 + c) * 129 + k] = b;
        D[(32 + c) * 129 + k] = 0.5f * (a + b);
    }
    __syncthreads();
    for (int idx = tid; idx < 48 * 128; idx += 256) {
        int r = idx >> 7, k = idx & 127;
        LG[r * 129 + k] = __logf(D[r * 129 + k]);
    }
    __syncthreads();

    // J = 0.5*(KL(PA) + KL(PB)), PA=[dA;M], PB=[dB;M], zero diagonal
    for (int p = tid; p < 32 * 32; p += 256) {
        int i = p >> 5, j = p & 31;
        float v = 0.f;
        if (i != j) {
            int rai = (i < 16) ? i : i + 16;      // dA row or M row
            int raj = (j < 16) ? j : j + 16;
            int rbi = i + 16;                     // dB row or M row
            int rbj = j + 16;
            const float* Pa  = &D[rai * 129];  const float* La = &LG[rai * 129];
            const float* Laj = &LG[raj * 129];
            const float* Pb  = &D[rbi * 129];  const float* Lb = &LG[rbi * 129];
            const float* Lbj = &LG[rbj * 129];
            float s = 0.f;
            for (int k = 0; k < NBINS; ++k)
                s += Pa[k] * (La[k] - Laj[k]) + Pb[k] * (Lb[k] - Lbj[k]);
            v = 0.5f * s;
        }
        Jm[i * 33 + j] = v;
    }
    __syncthreads();

    if (tid < 32) {
        int i = tid;
        float den = 0.f;
        for (int j = 0; j < 32; ++j) den += __expf(Jm[i * 33 + j] * 10.0f);  // /TAU
        float pos = (i < 16) ? Jm[i * 33 + (i + 16)] : Jm[i * 33 + (i - 16)];
        partial[i] = -pos * 10.0f + __logf(den);
    }
    __syncthreads();
    if (tid == 0) {
        float s = 0.f;
        for (int i = 0; i < 32; ++i) s += partial[i];
        out[0] = s * (1.0f / 32.0f);
    }
}

// ---------------------------------------------------------------------------
__global__ void clc_zero_kernel(float* __restrict__ ws, int n)
{
    for (int i = threadIdx.x; i < n; i += 256) ws[i] = 0.f;
}

// ---------------------------------------------------------------------------
extern "C" void kernel_launch(void* const* d_in, const int* in_sizes, int n_in,
                              void* d_out, int out_size, void* d_ws, size_t ws_size,
                              hipStream_t stream) {
    const float* x1   = (const float*)d_in[0];
    const int*   y1   = (const int*)  d_in[1];
    const float* x2   = (const float*)d_in[2];
    const int*   y2   = (const int*)  d_in[3];
    const float* W1   = (const float*)d_in[4];
    const float* b1   = (const float*)d_in[5];
    const float* W2   = (const float*)d_in[6];
    const float* b2   = (const float*)d_in[7];
    const float* bins = (const float*)d_in[8];

    float* ws    = (float*)d_ws;
    float* gsums = ws;          // 2*16*128 floats
    float* gcnt  = ws + 4096;   // 2*16 floats

    (void)hipFuncSetAttribute((const void*)clc_fused_kernel,
                              hipFuncAttributeMaxDynamicSharedMemorySize, DYN_LDS);

    clc_zero_kernel<<<1, 256, 0, stream>>>(ws, 4096 + 32);
    clc_fused_kernel<<<64, 256, DYN_LDS, stream>>>(x1, y1, x2, y2, W1, b1, W2, b2,
                                                   bins, gsums, gcnt);
    clc_finalize_kernel<<<1, 256, 0, stream>>>(gsums, gcnt, (float*)d_out);
}
